// ConvLSTM_8263517077900
// MI455X (gfx1250) — compile-verified, matrix operand order checked by reading, not run on hardware
//
#include <hip/hip_runtime.h>
#include <stddef.h>

typedef __attribute__((ext_vector_type(16))) __bf16 v16bf;
typedef __attribute__((ext_vector_type(8)))  float  v8f;

union Frag { unsigned int u[8]; v16bf v; };

static __device__ __forceinline__ unsigned short f2bf(float f) {
    unsigned int u = __float_as_uint(f);
    u = (u + 0x7FFFu + ((u >> 16) & 1u)) >> 16;   // round-to-nearest-even
    return (unsigned short)u;
}

// Async global->LDS 16B copy (gfx1250 GLOBAL_LOAD_ASYNC_TO_LDS_B128, ASYNCcnt).
static __device__ __forceinline__ void async_copy_b128(const void* gptr, void* lptr) {
    unsigned long long ga = (unsigned long long)(size_t)gptr;
    unsigned           la = (unsigned)(size_t)lptr;   // low 32 bits = LDS offset
    asm volatile("global_load_async_to_lds_b128 %0, %1, off"
                 :: "v"(la), "v"(ga) : "memory");
}
static __device__ __forceinline__ void wait_async0() {
    asm volatile("s_wait_asynccnt 0x0" ::: "memory");
}

// ---------------------------------------------------------------------------
// 1) Per-channel BN statistics: scale = gamma*rsqrt(var+eps), shift = beta-mean*scale
//    x viewed as [B=64][C=64][4096]
// ---------------------------------------------------------------------------
__global__ void bn_stats(const float* __restrict__ x,
                         const float* __restrict__ gamma,
                         const float* __restrict__ beta,
                         float* __restrict__ scale,
                         float* __restrict__ shift) {
    const int ci = blockIdx.x;          // 0..63
    const int tid = threadIdx.x;        // 256 threads
    float s = 0.f, q = 0.f;
    for (int b = 0; b < 64; ++b) {
        const float* p = x + ((size_t)(b * 64 + ci)) * 4096;
        for (int i = tid; i < 4096; i += 256) {
            float v = p[i];
            s += v;
            q += v * v;
        }
    }
    __shared__ float rs[256], rq[256];
    rs[tid] = s; rq[tid] = q;
    __syncthreads();
    for (int off = 128; off > 0; off >>= 1) {
        if (tid < off) { rs[tid] += rs[tid + off]; rq[tid] += rq[tid + off]; }
        __syncthreads();
    }
    if (tid == 0) {
        const float inv_n = 1.0f / 262144.0f;
        float mean = rs[0] * inv_n;
        float var  = rq[0] * inv_n - mean * mean;
        float sc   = gamma[ci] * rsqrtf(var + 1e-5f);
        scale[ci]  = sc;
        shift[ci]  = beta[ci] - mean * sc;
    }
}

// ---------------------------------------------------------------------------
// 2) Apply BN and convert NCHW fp32 -> NHWC bf16 (zero-padding then matches bnx)
// ---------------------------------------------------------------------------
__global__ void bn_convert(const float* __restrict__ x,
                           const float* __restrict__ scale,
                           const float* __restrict__ shift,
                           unsigned short* __restrict__ xb) {
    int id = blockIdx.x * 256 + threadIdx.x;       // NCHW linear index
    int w  = id & 63;
    int h  = (id >> 6) & 63;
    int ci = (id >> 12) & 63;
    int b  = id >> 18;
    float v = x[id] * scale[ci] + shift[ci];
    xb[(((size_t)(b * 64 + h) * 64) + w) * 64 + ci] = f2bf(v);
}

// ---------------------------------------------------------------------------
// 3a) Repack conv1_w [192][64][3][3] fp32 -> [tap(9)][co(192)][ci(64)] bf16
// ---------------------------------------------------------------------------
__global__ void pack_w1(const float* __restrict__ w, unsigned short* __restrict__ wb) {
    int id = blockIdx.x * 256 + threadIdx.x;       // 110592
    int ci = id & 63;
    int o  = (id >> 6) % 192;
    int t  = id / (64 * 192);
    wb[id] = f2bf(w[((size_t)(o * 64 + ci)) * 9 + t]);
}

// 3b) Pack [w_h2h | w_c2h] -> [tap(9)][co(192)][i2(128)] bf16 (i2<64: h, else c)
__global__ void pack_whc(const float* __restrict__ wh,
                         const float* __restrict__ wc,
                         unsigned short* __restrict__ wb) {
    int id = blockIdx.x * 256 + threadIdx.x;       // 221184
    int i2 = id & 127;
    int o  = (id >> 7) % 192;
    int t  = id / (128 * 192);
    float v = (i2 < 64) ? wh[((size_t)(o * 64 + i2)) * 9 + t]
                        : wc[((size_t)(o * 64 + (i2 - 64))) * 9 + t];
    wb[id] = f2bf(v);
}

// ---------------------------------------------------------------------------
// 4) 3x3 conv as implicit GEMM via v_wmma_f32_16x16x32_bf16.
//    Block: 192 threads = 6 waves; computes 16(pixels along W) x 192(ch) tile.
//    Grid: NB*64*4 blocks.  3x18xCin bf16 input patch staged in LDS via
//    GLOBAL_LOAD_ASYNC_TO_LDS_B128 (halo chunks zero-filled with ds stores).
// ---------------------------------------------------------------------------
__global__ void __launch_bounds__(192)
conv3x3_wmma(const unsigned short* __restrict__ xin,   // [NB][64][64][Cin] bf16
             const unsigned short* __restrict__ wbf,   // [9][192][Cin] bf16
             const float* __restrict__ bias,           // [192] or nullptr
             const float* __restrict__ addf,           // [NB][64][64][192] or nullptr
             float* __restrict__ out,                  // [NB][64][64][192]
             int Cin) {
    const int H = 64, W = 64, CO = 192;
    const int bid = blockIdx.x;
    const int wi = bid & 3;
    const int h  = (bid >> 2) & 63;
    const int n  = bid >> 8;
    const int w0 = wi * 16;
    const int tid  = threadIdx.x;
    const int wave = tid >> 5;
    const int lane = tid & 31;
    const int m  = lane & 15;      // row (pixel) for A / column for B,D
    const int hi = lane >> 4;      // lane half -> K sub-block / D row half
    const int Cin2 = Cin >> 1;

    __shared__ __align__(16) unsigned int patch[3 * 18 * 64];  // up to Cin=128
    const unsigned int* xinU = (const unsigned int*)xin;

    // Stage 3x18 pixel patch (all channels) into LDS in 16B chunks.
    const int CH4 = Cin2 >> 2;           // b128 chunks per pixel (8 or 16)
    const int nchunk = 54 * CH4;         // 432 or 864
    for (int j = tid; j < nchunk; j += 192) {
        int q  = j % CH4;                // 16B chunk within pixel
        int pc = j / CH4;                // pixel index: r*18 + cc
        int cc = pc % 18;
        int r  = pc / 18;
        int ih = h - 1 + r;
        int iw = w0 - 1 + cc;
        unsigned int* lp = &patch[pc * Cin2 + q * 4];
        if (ih >= 0 && ih < H && iw >= 0 && iw < W) {
            const unsigned int* gp =
                xinU + ((size_t)((n * H + ih) * W + iw)) * Cin2 + q * 4;
            async_copy_b128(gp, lp);
        } else {
            lp[0] = 0; lp[1] = 0; lp[2] = 0; lp[3] = 0;   // zero padding
        }
    }
    wait_async0();
    __syncthreads();

    v8f acc0 = {};
    v8f acc1 = {};
    const unsigned int* wU = (const unsigned int*)wbf;
    const int co0 = wave * 32 + m;
    const int co1 = co0 + 16;

    for (int t = 0; t < 9; ++t) {
        const int kh = t / 3, kw = t - 3 * kh;
        const int arow = (kh * 18 + (m + kw)) * Cin2;
        for (int k0 = 0; k0 < Cin; k0 += 32) {
            Frag A, B0, B1;
#pragma unroll
            for (int v = 0; v < 8; ++v) {
                // 16-bit A 16x32 / B 32x16 lane-VGPR swizzle (ISA 7.12.2)
                int k = k0 + hi * 8 + ((v < 4) ? (v * 2) : (16 + (v - 4) * 2));
                A.u[v]  = patch[arow + (k >> 1)];
                B0.u[v] = wU[((size_t)(t * CO + co0)) * Cin2 + (k >> 1)];
                B1.u[v] = wU[((size_t)(t * CO + co1)) * Cin2 + (k >> 1)];
            }
            acc0 = __builtin_amdgcn_wmma_f32_16x16x32_bf16(
                false, A.v, false, B0.v, (short)0, acc0, false, false);
            acc1 = __builtin_amdgcn_wmma_f32_16x16x32_bf16(
                false, A.v, false, B1.v, (short)0, acc1, false, false);
        }
    }

    // Epilogue: D layout — lane holds col (lane&15), rows v + 8*(lane>>4)
    const size_t base = ((size_t)(n * H + h) * W + w0);
    const float bv0 = bias ? bias[co0] : 0.f;
    const float bv1 = bias ? bias[co1] : 0.f;
#pragma unroll
    for (int v = 0; v < 8; ++v) {
        int mm = v + hi * 8;
        size_t idx = (base + mm) * CO;
        float a0 = addf ? addf[idx + co0] : bv0;
        float a1 = addf ? addf[idx + co1] : bv1;
        out[idx + co0] = acc0[v] + a0;
        out[idx + co1] = acc1[v] + a1;
    }
}

// ---------------------------------------------------------------------------
// 5) LSTM pointwise: gates, state update, emit bf16 [h|c] for next GEMM,
//    and h into d_out (NCHW per step).  1,048,576 elements.
// ---------------------------------------------------------------------------
__global__ void lstm_point(const float* __restrict__ pre,       // [4][64][64][192]
                           float* __restrict__ c_state,         // [4][64][64][64]
                           unsigned short* __restrict__ hc,     // [4][64][64][128]
                           float* __restrict__ out_t) {         // [4][64][4096]
    int id = blockIdx.x * 256 + threadIdx.x;
    int co = id & 63;
    int w  = (id >> 6) & 63;
    int h  = (id >> 12) & 63;
    int n  = id >> 18;
    size_t p = ((size_t)(n * 64 + h) * 64) + w;
    float f = pre[p * 192 + co];
    float o = pre[p * 192 + 64 + co];
    float g = pre[p * 192 + 128 + co];
    f = 1.f / (1.f + expf(-f));
    o = 1.f / (1.f + expf(-o));
    g = tanhf(g);
    float c_old = c_state[p * 64 + co];
    float c_new = f * c_old + (1.f - f) * g;
    float h_new = o * tanhf(c_new);
    c_state[p * 64 + co] = c_new;
    hc[p * 128 + co]       = f2bf(h_new);
    hc[p * 128 + 64 + co]  = f2bf(c_new);
    out_t[((size_t)(n * 64 + co)) * 4096 + h * 64 + w] = h_new;
}

// ---------------------------------------------------------------------------
extern "C" void kernel_launch(void* const* d_in, const int* in_sizes, int n_in,
                              void* d_out, int out_size, void* d_ws, size_t ws_size,
                              hipStream_t stream) {
    const float* x       = (const float*)d_in[0];   // [16][4][64][64][64]
    const float* gamma   = (const float*)d_in[1];   // [64]
    const float* beta    = (const float*)d_in[2];   // [64]
    const float* conv1_w = (const float*)d_in[3];   // [192][64][3][3]
    const float* conv1_b = (const float*)d_in[4];   // [192]
    const float* w_h2h   = (const float*)d_in[5];   // [192][64][3][3]
    const float* w_c2h   = (const float*)d_in[6];   // [192][64][3][3]
    float* out = (float*)d_out;                     // [16][4][64][64][64]

    char* ws = (char*)d_ws;
    size_t off = 0;
    unsigned short* xb_bf  = (unsigned short*)(ws + off); off += 64ull*4096*64*2;    // 33.5 MB
    unsigned short* w1_bf  = (unsigned short*)(ws + off); off += 9ull*192*64*2;      // 216 KB
    unsigned short* whc_bf = (unsigned short*)(ws + off); off += 9ull*192*128*2;     // 432 KB
    float* scale  = (float*)(ws + off); off += 256;
    float* shift  = (float*)(ws + off); off += 256;
    float* y      = (float*)(ws + off); off += 64ull*4096*192*4;                     // 201 MB
    float* preact = (float*)(ws + off); off += 4ull*4096*192*4;                      // 12.6 MB
    float* c_st   = (float*)(ws + off); off += 4ull*4096*64*4;                       // 4.2 MB
    unsigned short* hc_bf = (unsigned short*)(ws + off); off += 4ull*4096*128*2;     // 4.2 MB

    // zero-init recurrent state (h0 = c0 = 0)
    hipMemsetAsync(c_st,  0, 4ull*4096*64*4,  stream);
    hipMemsetAsync(hc_bf, 0, 4ull*4096*128*2, stream);

    bn_stats<<<64, 256, 0, stream>>>(x, gamma, beta, scale, shift);
    bn_convert<<<65536, 256, 0, stream>>>(x, scale, shift, xb_bf);
    pack_w1<<<432, 256, 0, stream>>>(conv1_w, w1_bf);
    pack_whc<<<864, 256, 0, stream>>>(w_h2h, w_c2h, whc_bf);

    // conv1 over all T*N = 64 images: y = conv(bnx) + bias   (implicit GEMM, WMMA)
    conv3x3_wmma<<<64 * 64 * 4, 192, 0, stream>>>(
        xb_bf, w1_bf, conv1_b, nullptr, y, 64);

    // 16 serial LSTM steps: fused [h|c] GEMM (K = 9*128) + pointwise gates
    for (int t = 0; t < 16; ++t) {
        const float* y_t = y + (size_t)t * 4 * 4096 * 192;
        conv3x3_wmma<<<4 * 64 * 4, 192, 0, stream>>>(
            hc_bf, whc_bf, nullptr, y_t, preact, 128);
        lstm_point<<<4096, 256, 0, stream>>>(
            preact, c_st, hc_bf, out + (size_t)t * 4 * 64 * 4096);
    }
    (void)in_sizes; (void)n_in; (void)out_size; (void)ws_size;
}
